// MatrixSparseCodingLayer_55044300865755
// MI455X (gfx1250) — matrix operand order repaired and checked by reading, not compiled, not run on hardware
//
#include <hip/hip_runtime.h>
#include <math.h>

// ---------------------------------------------------------------------------
// Problem constants (from the reference)
// ---------------------------------------------------------------------------
#define N_ROWS 16384   // 8 * 2048 flattened tokens
#define C_DIM  512     // channels
#define D_DIM  2048    // dictionary atoms
#define STEP_S 0.1f
#define LMBD_F 0.1f
#define LAM_F  0.01f   // lmbd * step (mu = 0 -> s_mu = 1)

typedef __bf16 bf16_t;
typedef __bf16 v16bf __attribute__((ext_vector_type(16)));
typedef __bf16 v8bf  __attribute__((ext_vector_type(8)));
typedef float  v8f   __attribute__((ext_vector_type(8)));

union Frag16 { v16bf v; v8bf h[2]; };

// ---------------------------------------------------------------------------
// Input conversion: x -> bf16, W -> bf16 (row-major) and bf16 transposed
// ---------------------------------------------------------------------------
__global__ void convert_kernel(const float* __restrict__ x,
                               const float* __restrict__ W,
                               bf16_t* __restrict__ xbf,
                               bf16_t* __restrict__ wbf,
                               bf16_t* __restrict__ wtbf,
                               long nx, long nw) {
  long stride = (long)gridDim.x * blockDim.x;
  for (long i = (long)blockIdx.x * blockDim.x + threadIdx.x; i < nx; i += stride) {
    xbf[i] = (bf16_t)x[i];
    if (i < nw) {
      float w = W[i];
      wbf[i] = (bf16_t)w;
      long d = i >> 9;          // / 512
      long cc = i & 511;        // % 512
      wtbf[cc * (long)D_DIM + d] = (bf16_t)w;
    }
  }
}

// ---------------------------------------------------------------------------
// Shared 256x128 tile GEMM core (bf16 WMMA, f32 accumulate).
// A: [M,K] row-major (lda), B: [Nd,K] row-major (ldb)  => Dout = A * B^T.
// 8 waves arranged 4(M) x 2(N); each wave owns a 64x64 patch = 4x4 WMMA
// tiles, giving a 1:1 WMMA : ds_load_b128 ratio in the hot loop.
// Staging is load-all-then-store-all (no control flow in between) so the
// global loads batch under one loadcnt wait; A-rows get a uniform, branch-
// free GL2 prefetch (global_prefetch_b8, SCOPE_DEV) 2 chunks ahead.
// ---------------------------------------------------------------------------
__device__ __forceinline__ void gemm_tile_core(
    const bf16_t* __restrict__ A, const bf16_t* __restrict__ B,
    int lda, int ldb, int K,
    bf16_t (&As)[256][32], bf16_t (&Bs)[128][32],
    v8f acc[4][4]) {
  const int tid  = threadIdx.x;
  const int lane = tid & 31;
  const int wid  = tid >> 5;
  const int wm   = wid >> 1;   // 0..3 : 64 rows each
  const int wn   = wid & 1;    // 0..1 : 64 cols each
  const long rowBase = (long)blockIdx.x * 256;
  const long colBase = (long)blockIdx.y * 128;

  v8f z = {};
#pragma unroll
  for (int i = 0; i < 4; ++i)
#pragma unroll
    for (int j = 0; j < 4; ++j) acc[i][j] = z;

  const int hsel = lane >> 4;  // half-wave select
  const int lr   = lane & 15;

  for (int kb = 0; kb < K; kb += 32) {
    // Uniform prefetch: thread t covers A row (rowBase + t), 2 chunks ahead.
    // Scalar guard -> no EXEC divergence, loads below stay batchable.
    if (kb + 64 < K)
      __builtin_prefetch(&A[(rowBase + tid) * (long)lda + kb + 64], 0, 1);

    // Batched global loads into registers (6 x b128 in flight).
    v8bf ta[4], tb[2];
#pragma unroll
    for (int it = 0; it < 4; ++it) {
      int idx = tid + it * 256;          // 0..1023
      int r   = idx >> 2;
      int ch  = idx & 3;
      ta[it] = *(const v8bf*)&A[(rowBase + r) * (long)lda + kb + ch * 8];
    }
#pragma unroll
    for (int it = 0; it < 2; ++it) {
      int idx = tid + it * 256;          // 0..511
      int r   = idx >> 2;
      int ch  = idx & 3;
      tb[it] = *(const v8bf*)&B[(colBase + r) * (long)ldb + kb + ch * 8];
    }
    // Store to LDS: A 256x32 (16 KB), B 128x32 (8 KB).
#pragma unroll
    for (int it = 0; it < 4; ++it) {
      int idx = tid + it * 256;
      int r   = idx >> 2;
      int ch  = idx & 3;
      *(v8bf*)&As[r][ch * 8] = ta[it];
    }
#pragma unroll
    for (int it = 0; it < 2; ++it) {
      int idx = tid + it * 256;
      int r   = idx >> 2;
      int ch  = idx & 3;
      *(v8bf*)&Bs[r][ch * 8] = tb[it];
    }
    __syncthreads();

    // Fragment gather per ISA 7.12.2 bf16 layouts.
    // A 16x32: lanes 0-15 hold K {0..7,16..23}; lanes 16-31 hold {8..15,24..31}.
    // B 32x16: same K order as A: lanes 0-15 hold K {0..7,16..23}; lanes 16-31 hold {8..15,24..31}.
    Frag16 a[4], b[4];
#pragma unroll
    for (int tm = 0; tm < 4; ++tm) {
      const v8bf* p = (const v8bf*)&As[wm * 64 + tm * 16 + lr][0];
      a[tm].h[0] = p[hsel];
      a[tm].h[1] = p[hsel + 2];
    }
#pragma unroll
    for (int tn = 0; tn < 4; ++tn) {
      const v8bf* p = (const v8bf*)&Bs[wn * 64 + tn * 16 + lr][0];
      b[tn].h[0] = p[hsel];
      b[tn].h[1] = p[hsel + 2];
    }

#pragma unroll
    for (int tm = 0; tm < 4; ++tm)
#pragma unroll
      for (int tn = 0; tn < 4; ++tn)
        acc[tm][tn] = __builtin_amdgcn_wmma_f32_16x16x32_bf16(
            false, a[tm].v, false, b[tn].v, (short)0, acc[tm][tn],
            false, false);
    __syncthreads();
  }
}

// ---------------------------------------------------------------------------
// GEMM 1: G = r @ W^T  [N_ROWS x D_DIM], fused FISTA update epilogue:
//   a = alpha*c + beta*c_pre ; c_pre = c ; c = relu(softshrink(a + s*G))
//   relu(softshrink(v)) == max(v - lam, 0)
// ---------------------------------------------------------------------------
__global__ __launch_bounds__(256)
void fista_update_kernel(const bf16_t* __restrict__ Abf,  // [N_ROWS, 512]
                         const bf16_t* __restrict__ Wbf,  // [2048, 512]
                         float* __restrict__ c,           // [N_ROWS, 2048]
                         float* __restrict__ c_pre,
                         bf16_t* __restrict__ cbf,
                         int mode, float alpha, float beta) {
  __shared__ bf16_t As[256][32];
  __shared__ bf16_t Bs[128][32];
  v8f acc[4][4];
  gemm_tile_core(Abf, Wbf, C_DIM, C_DIM, C_DIM, As, Bs, acc);

  const int lane = threadIdx.x & 31;
  const int wid  = threadIdx.x >> 5;
  const int wm = wid >> 1, wn = wid & 1;
  const long rowBase = (long)blockIdx.x * 256 + wm * 64 + 8 * (lane >> 4);
  const long colBase = (long)blockIdx.y * 128 + wn * 64 + (lane & 15);

#pragma unroll
  for (int tm = 0; tm < 4; ++tm)
#pragma unroll
    for (int tn = 0; tn < 4; ++tn)
#pragma unroll
      for (int e = 0; e < 8; ++e) {
        long row = rowBase + tm * 16 + e;
        long col = colBase + tn * 16;
        long idx = row * (long)D_DIM + col;
        float g = acc[tm][tn][e];
        float aval = 0.f;
        if (mode != 0) {
          float cold = c[idx];
          aval = alpha * cold;
          if (beta != 0.f) aval += beta * c_pre[idx];
          c_pre[idx] = cold;
        }
        float v  = aval + STEP_S * g;
        float cn = fmaxf(v - LAM_F, 0.f);
        c[idx]   = cn;
        cbf[idx] = (bf16_t)cn;
      }
}

// ---------------------------------------------------------------------------
// GEMM 2: P = c @ W  [N_ROWS x C_DIM], fused residual epilogue:
//   r = x - P ; non-final: write rbf ; final: write xp, r, per-block sum(r^2)
// ---------------------------------------------------------------------------
__global__ __launch_bounds__(256)
void residual_kernel(const bf16_t* __restrict__ cbf,   // [N_ROWS, 2048]
                     const bf16_t* __restrict__ wtbf,  // [512, 2048]
                     const float* __restrict__ x,      // [N_ROWS, 512]
                     bf16_t* __restrict__ rbf,
                     float* __restrict__ xp_out,
                     float* __restrict__ r_out,
                     float* __restrict__ partials,
                     int final_flag) {
  __shared__ bf16_t As[256][32];
  __shared__ bf16_t Bs[128][32];
  __shared__ float red[256];
  v8f acc[4][4];
  gemm_tile_core(cbf, wtbf, D_DIM, D_DIM, D_DIM, As, Bs, acc);

  const int lane = threadIdx.x & 31;
  const int wid  = threadIdx.x >> 5;
  const int wm = wid >> 1, wn = wid & 1;
  const long rowBase = (long)blockIdx.x * 256 + wm * 64 + 8 * (lane >> 4);
  const long colBase = (long)blockIdx.y * 128 + wn * 64 + (lane & 15);

  float r2 = 0.f;
#pragma unroll
  for (int tm = 0; tm < 4; ++tm)
#pragma unroll
    for (int tn = 0; tn < 4; ++tn)
#pragma unroll
      for (int e = 0; e < 8; ++e) {
        long row = rowBase + tm * 16 + e;
        long col = colBase + tn * 16;
        long idx = row * (long)C_DIM + col;
        float p  = acc[tm][tn][e];
        float rv = x[idx] - p;
        if (final_flag) {
          xp_out[idx] = p;
          r_out[idx]  = rv;
          r2 += rv * rv;
        } else {
          rbf[idx] = (bf16_t)rv;
        }
      }

  if (final_flag) {
    red[threadIdx.x] = r2;
    __syncthreads();
    for (int off = 128; off > 0; off >>= 1) {
      if ((int)threadIdx.x < off) red[threadIdx.x] += red[threadIdx.x + off];
      __syncthreads();
    }
    if (threadIdx.x == 0)
      partials[blockIdx.y * gridDim.x + blockIdx.x] = red[0];
  }
}

// ---------------------------------------------------------------------------
// c_loss partials: per-block sum(|c|)
// ---------------------------------------------------------------------------
__global__ __launch_bounds__(256)
void abs_sum_kernel(const float* __restrict__ c, float* __restrict__ partC,
                    long n) {
  __shared__ float red[256];
  float s = 0.f;
  long stride = (long)gridDim.x * blockDim.x;
  for (long i = (long)blockIdx.x * blockDim.x + threadIdx.x; i < n; i += stride)
    s += fabsf(c[i]);
  red[threadIdx.x] = s;
  __syncthreads();
  for (int off = 128; off > 0; off >>= 1) {
    if ((int)threadIdx.x < off) red[threadIdx.x] += red[threadIdx.x + off];
    __syncthreads();
  }
  if (threadIdx.x == 0) partC[blockIdx.x] = red[0];
}

// ---------------------------------------------------------------------------
// Deterministic final reduction of both loss partial arrays.
// ---------------------------------------------------------------------------
__global__ __launch_bounds__(256)
void finalize_kernel(const float* __restrict__ partA, int nA,
                     const float* __restrict__ partC, int nC,
                     float* __restrict__ rloss, float* __restrict__ closs) {
  __shared__ float red[256];
  float s = 0.f;
  for (int i = threadIdx.x; i < nA; i += 256) s += partA[i];
  red[threadIdx.x] = s;
  __syncthreads();
  for (int off = 128; off > 0; off >>= 1) {
    if ((int)threadIdx.x < off) red[threadIdx.x] += red[threadIdx.x + off];
    __syncthreads();
  }
  if (threadIdx.x == 0) *rloss = red[0];   // / N_DICT == 1
  __syncthreads();

  s = 0.f;
  for (int i = threadIdx.x; i < nC; i += 256) s += partC[i];
  red[threadIdx.x] = s;
  __syncthreads();
  for (int off = 128; off > 0; off >>= 1) {
    if ((int)threadIdx.x < off) red[threadIdx.x] += red[threadIdx.x + off];
    __syncthreads();
  }
  if (threadIdx.x == 0) *closs = LMBD_F * red[0];  // mu == 0
}

// ---------------------------------------------------------------------------
// Host orchestration
// ---------------------------------------------------------------------------
extern "C" void kernel_launch(void* const* d_in, const int* in_sizes, int n_in,
                              void* d_out, int out_size, void* d_ws,
                              size_t ws_size, hipStream_t stream) {
  (void)in_sizes; (void)n_in; (void)out_size; (void)ws_size;

  const float* x = (const float*)d_in[0];
  const float* W = (const float*)d_in[1];

  float* out    = (float*)d_out;
  float* c_out  = out;                               // 16384*2048
  float* xp_out = out + (long)N_ROWS * D_DIM;        // 16384*512
  float* r_out  = xp_out + (long)N_ROWS * C_DIM;     // 16384*512
  float* rloss  = r_out + (long)N_ROWS * C_DIM;
  float* closs  = rloss + 1;

  char* ws = (char*)d_ws;
  bf16_t* xbf   = (bf16_t*)(ws);                     //  16 MB
  bf16_t* wbf   = (bf16_t*)(ws + 16777216);          //   2 MB
  bf16_t* wtbf  = (bf16_t*)(ws + 18874368);          //   2 MB
  bf16_t* cbf   = (bf16_t*)(ws + 20971520);          //  64 MB
  bf16_t* rbf   = (bf16_t*)(ws + 88080384);          //  16 MB
  float*  cpre  = (float*)(ws + 104857600);          // 128 MB
  float*  partA = (float*)(ws + 239075328);          // 256 floats
  float*  partC = (float*)(ws + 239077376);          // 1024 floats

  // FISTA momentum coefficients (t-sequence), i = 1..4
  float alpha[5] = {0.f, 1.f, 0.f, 0.f, 0.f};
  float beta[5]  = {0.f, 0.f, 0.f, 0.f, 0.f};
  double t = (sqrt(5.0) + 1.0) / 2.0;
  for (int i = 2; i < 5; ++i) {
    double tp = t;
    t = (sqrt(1.0 + 4.0 * tp * tp) + 1.0) / 2.0;
    alpha[i] = (float)((tp + t - 1.0) / t);
    beta[i]  = (float)((1.0 - tp) / t);
  }

  convert_kernel<<<4096, 256, 0, stream>>>(
      x, W, xbf, wbf, wtbf, (long)N_ROWS * C_DIM, (long)D_DIM * C_DIM);

  dim3 gUpd(N_ROWS / 256, D_DIM / 128);  // 64 x 16
  dim3 gRes(N_ROWS / 256, C_DIM / 128);  // 64 x 4

  // i == 0 : c = relu(prox(s * x @ W^T))
  fista_update_kernel<<<gUpd, 256, 0, stream>>>(xbf, wbf, c_out, cpre, cbf,
                                                0, 0.f, 0.f);
  // i == 1..4
  for (int i = 1; i < 5; ++i) {
    residual_kernel<<<gRes, 256, 0, stream>>>(cbf, wtbf, x, rbf,
                                              nullptr, nullptr, nullptr, 0);
    fista_update_kernel<<<gUpd, 256, 0, stream>>>(rbf, wbf, c_out, cpre, cbf,
                                                  1, alpha[i], beta[i]);
  }

  // Final reconstruction: xp, r, and r^2 partials
  residual_kernel<<<gRes, 256, 0, stream>>>(cbf, wtbf, x, rbf,
                                            xp_out, r_out, partA, 1);

  abs_sum_kernel<<<1024, 256, 0, stream>>>(c_out, partC,
                                           (long)N_ROWS * D_DIM);
  finalize_kernel<<<1, 256, 0, stream>>>(partA, gRes.x * gRes.y,
                                         partC, 1024, rloss, closs);
}
